// NaryLSTMLayer_13683765805741
// MI455X (gfx1250) — matrix operand order repaired and checked by reading, not compiled, not run on hardware
//
#include <hip/hip_runtime.h>
#include <hip/hip_bf16.h>

// CDNA5 (gfx1250) wave32 WMMA types
typedef __attribute__((ext_vector_type(16))) __bf16 v16bf;
typedef __attribute__((ext_vector_type(8)))  float  v8f;

#define T_STEPS 64
#define NN      8192
#define DD      128

// Per-jt packed weight block: 56 fragments x (32 lanes x 16 bf16) = 28672 bf16 = 56 KB
#define FRAGS_PER_JT   56
#define BF16_PER_JT    (FRAGS_PER_JT * 32 * 16)   // 28672
#define UINT4_PER_JT   (BF16_PER_JT / 8)          // 3584
#define TOTAL_PACKED   (8 * BF16_PER_JT)          // 229376 bf16 = 448 KB

// ---------------- helpers ----------------

__device__ __forceinline__ unsigned short f2bf_rne(float f) {
  unsigned u = __float_as_uint(f);
  u += 0x7FFFu + ((u >> 16) & 1u);     // round-to-nearest-even
  return (unsigned short)(u >> 16);
}

// fast transcendental path: v_exp_f32 + v_rcp_f32 (no IEEE div expansion)
__device__ __forceinline__ float fexp2_(float x) { return __builtin_amdgcn_exp2f(x); }
__device__ __forceinline__ float sigmoidf_(float x) {
  return __builtin_amdgcn_rcpf(1.0f + fexp2_(x * -1.4426950408889634f));
}
__device__ __forceinline__ float tanhf_(float x) {
  x = fminf(fmaxf(x, -15.0f), 15.0f);
  // tanh(x) = 1 - 2/(exp(2x)+1)
  return 1.0f - 2.0f * __builtin_amdgcn_rcpf(fexp2_(x * 2.8853900817779268f) + 1.0f);
}

// Build a 16x32 bf16 A-fragment for this lane from a row-major f32 row.
// ISA 7.12.2 (16-bit A, wave32): lane m=l&15, hi=l>>4;
//   dword j in 0..3 holds K = kbase + 2j + 8*hi (pair)
//   dword j in 4..7 holds K = kbase + 16 + 2(j-4) + 8*hi (pair)
__device__ __forceinline__ v16bf a_frag_from_row(const float* row, int kbase, int hi) {
  union { v16bf v; unsigned u32[8]; } r;
  if (row) {
#pragma unroll
    for (int j = 0; j < 8; ++j) {
      int k = kbase + ((j < 4) ? (2 * j + 8 * hi) : (16 + 2 * (j - 4) + 8 * hi));
      float2 p = *(const float2*)(row + k);           // k even -> 8B aligned
      unsigned lo   = __float_as_uint(p.x) >> 16;     // truncate to bf16
      unsigned hi16 = __float_as_uint(p.y) >> 16;
      r.u32[j] = lo | (hi16 << 16);
    }
  } else {
#pragma unroll
    for (int j = 0; j < 8; ++j) r.u32[j] = 0u;
  }
  return r.v;
}

// Load one B fragment (32 lanes x 16 bf16) from LDS (lane-contiguous 32B)
__device__ __forceinline__ v16bf b_frag_lds(const unsigned short* B, int frag, int lane) {
  const uint4* p = (const uint4*)(B + ((size_t)(frag * 32 + lane)) * 16);
  union { v16bf v; uint4 q[2]; } r;
  r.q[0] = p[0];
  r.q[1] = p[1];
  return r.v;
}

__device__ __forceinline__ v8f wmma_bf16(v16bf a, v16bf b, v8f c) {
  return __builtin_amdgcn_wmma_f32_16x16x32_bf16(false, a, false, b, (short)0, c, false, false);
}

// ---------------- one-time weight pack ----------------
// Layout: 8 jt-blocks; each block = 56 fragments:
//   frag 0..15 : W_W  -> gate g = frag>>2, k-step q = frag&3, cols g*128 + jt*16 ..
//   frag 16..55: Ucat -> block b = (frag-16)>>3, q = (frag-16)&7, cols b*128 + jt*16 ..
// B-fragment element mapping (ISA 7.12.2, 32x16 16-bit B): lane l -> n=l&15, hi=l>>4;
//   packed u16 pos = 2j+w  ->  K = q*32 + (j<4 ? 8*hi + 2j : 16 + 8*hi + 2(j-4)) + w
__global__ void pack_weights_kernel(const float* __restrict__ W_W,
                                    const float* __restrict__ U_f1,
                                    const float* __restrict__ U_f2,
                                    const float* __restrict__ U_iuo,
                                    unsigned short* __restrict__ packed) {
  int e = blockIdx.x * blockDim.x + threadIdx.x;
  if (e >= TOTAL_PACKED) return;
  int jtb  = e / BF16_PER_JT;
  int r    = e - jtb * BF16_PER_JT;
  int frag = r >> 9;              // / (32*16)
  int lane = (r >> 4) & 31;
  int pos  = r & 15;
  int n = lane & 15, hi = lane >> 4;
  int j = pos >> 1, w = pos & 1;
  int kin = ((j < 4) ? (8 * hi + 2 * j) : (16 + 8 * hi + 2 * (j - 4))) + w;  // K within the 32-wide k-step
  float v;
  if (frag < 16) {
    int g = frag >> 2, q = frag & 3;
    int k = q * 32 + kin;
    v = W_W[k * 512 + g * 128 + jtb * 16 + n];
  } else {
    int f2 = frag - 16;
    int b = f2 >> 3, q = f2 & 7;
    int k = q * 32 + kin;
    int c = b * 128 + jtb * 16 + n;
    v = (c < 128) ? U_f1[k * 128 + c]
      : (c < 256) ? U_f2[k * 128 + (c - 128)]
                  : U_iuo[k * 384 + (c - 256)];
  }
  packed[e] = f2bf_rne(v);
}

// ---------------- per-step fused kernel ----------------
// Grid: 32 blocks x 512 threads (16 waves). Each wave owns a 16-node tile.
// B fragments are staged in double-buffered LDS (2 x 56 KB), shared by all 16 waves.
__global__ __launch_bounds__(512)
void tree_lstm_step_kernel(const float* __restrict__ x,        // [N,128] this step
                           const int*   __restrict__ idx,      // [N,2]   this step
                           const unsigned short* __restrict__ packed,
                           const float* __restrict__ bW,       // [512]
                           const float* __restrict__ prevH,    // [N,128] or null (t==0)
                           const float* __restrict__ prevC,    // [N,128] or null
                           float* __restrict__ outH,           // [N,128]
                           float* __restrict__ outC) {         // [N,128]
  __shared__ uint4 ldsB[2][UINT4_PER_JT];                      // 112 KB (<320 KB/WGP)

  const int tid  = threadIdx.x;
  const int lane = tid & 31;
  const int wave = tid >> 5;
  const int tile = blockIdx.x * 16 + wave;
  const int nodeBase = tile * 16;

  const int m   = lane & 15;   // A-side row for this lane
  const int hi  = lane >> 4;   // A-side K-half selector
  const int n   = lane & 15;   // C/D column for this lane
  const int hi2 = lane >> 4;   // C/D row-half selector
  (void)m;

  // ---- gather row pointers for this lane's A row (node nodeBase+m) ----
  const int node_m = nodeBase + (lane & 15);
  int i0 = idx[node_m * 2 + 0];
  int i1 = idx[node_m * 2 + 1];
  const float* h0 = (prevH && i0 > 0) ? (prevH + (size_t)(i0 - 1) * DD) : nullptr;
  const float* h1 = (prevH && i1 > 0) ? (prevH + (size_t)(i1 - 1) * DD) : nullptr;

  // ---- A fragments: x (K=128 -> 4 frags), hc (K=256 -> 8 frags) ----
  const float* xrow = x + (size_t)node_m * DD;
  v16bf ax[4];
#pragma unroll
  for (int q = 0; q < 4; ++q) ax[q] = a_frag_from_row(xrow, q * 32, hi);

  v16bf ahc[8];
#pragma unroll
  for (int q = 0; q < 4; ++q) ahc[q] = a_frag_from_row(h0, q * 32, hi);
#pragma unroll
  for (int q = 0; q < 4; ++q) ahc[4 + q] = a_frag_from_row(h1, q * 32, hi);

  // ---- gathered-c offsets for this lane's 8 C/D rows (m = r + 8*hi2) ----
  int offc0[8], offc1[8];
#pragma unroll
  for (int r = 0; r < 8; ++r) {
    int mr = nodeBase + r + 8 * hi2;
    int j0 = idx[mr * 2 + 0];
    int j1 = idx[mr * 2 + 1];
    offc0[r] = (prevC && j0 > 0) ? (j0 - 1) * DD : -1;
    offc1[r] = (prevC && j1 > 0) ? (j1 - 1) * DD : -1;
  }

  // ---- prologue: stage jt=0's 56 KB of B fragments into buffer 0 ----
  {
    const uint4* g = (const uint4*)packed;  // jt block 0
    uint4 tmp[7];
#pragma unroll
    for (int i = 0; i < 7; ++i) tmp[i] = g[tid + i * 512];
#pragma unroll
    for (int i = 0; i < 7; ++i) ldsB[0][tid + i * 512] = tmp[i];
  }
  __syncthreads();

  // ---- loop over 8 output column tiles of D=128 ----
#pragma unroll 1
  for (int jt = 0; jt < 8; ++jt) {
    // prefetch next jt's B block into the other LDS buffer
    if (jt < 7) {
      const uint4* g = (const uint4*)(packed + (size_t)(jt + 1) * BF16_PER_JT);
      uint4 tmp[7];
#pragma unroll
      for (int i = 0; i < 7; ++i) tmp[i] = g[tid + i * 512];
#pragma unroll
      for (int i = 0; i < 7; ++i) ldsB[(jt + 1) & 1][tid + i * 512] = tmp[i];
    }

    const unsigned short* B = (const unsigned short*)&ldsB[jt & 1][0];

    const int col = jt * 16 + n;
    const float bf = bW[  0 + col];
    const float bi = bW[128 + col];
    const float bu = bW[256 + col];
    const float bo = bW[384 + col];

    // gemm over x with W_W gate g (frags g*4 + q)
    auto gemmW = [&](int g) -> v8f {
      v8f acc = {};
#pragma unroll
      for (int q = 0; q < 4; ++q)
        acc = wmma_bf16(ax[q], b_frag_lds(B, g * 4 + q, lane), acc);
      return acc;
    };
    // gemm over hc with U block b (frags 16 + b*8 + q)
    auto gemmU = [&](int b) -> v8f {
      v8f acc = {};
#pragma unroll
      for (int q = 0; q < 8; ++q)
        acc = wmma_bf16(ahc[q], b_frag_lds(B, 16 + b * 8 + q, lane), acc);
      return acc;
    };

    // f1 / f2 / fc
    v8f wf = gemmW(0);
    v8f g0 = gemmU(0);
    v8f f1;
#pragma unroll
    for (int r = 0; r < 8; ++r) f1[r] = sigmoidf_(wf[r] + bf + g0[r]);
    v8f g1 = gemmU(1);
    v8f fc;
#pragma unroll
    for (int r = 0; r < 8; ++r) {
      float f2v = sigmoidf_(wf[r] + bf + g1[r]);
      float c0 = (offc0[r] >= 0) ? prevC[offc0[r] + col] : 0.0f;
      float c1 = (offc1[r] >= 0) ? prevC[offc1[r] + col] : 0.0f;
      fc[r] = f1[r] * c0 + f2v * c1;
    }

    // i * u
    v8f wi = gemmW(1);
    v8f g2 = gemmU(2);
    v8f wu = gemmW(2);
    v8f g3 = gemmU(3);
    v8f nc;
#pragma unroll
    for (int r = 0; r < 8; ++r) {
      float iv = sigmoidf_(wi[r] + bi + g2[r]);
      float uv = tanhf_(wu[r] + bu + g3[r]);
      nc[r] = iv * uv + fc[r];
    }

    // o, outputs
    v8f wo = gemmW(3);
    v8f g4 = gemmU(4);
#pragma unroll
    for (int r = 0; r < 8; ++r) {
      float ov = sigmoidf_(wo[r] + bo + g4[r]);
      float nh = ov * tanhf_(nc[r]);
      size_t rowOff = (size_t)(nodeBase + r + 8 * hi2) * DD;
      outC[rowOff + col] = nc[r];
      outH[rowOff + col] = nh;
    }

    __syncthreads();   // buffer rotation
  }
}

// ---------------- launch ----------------
extern "C" void kernel_launch(void* const* d_in, const int* in_sizes, int n_in,
                              void* d_out, int out_size, void* d_ws, size_t ws_size,
                              hipStream_t stream) {
  const float* tensor  = (const float*)d_in[0];   // [T,N,128]
  const int*   indices = (const int*)  d_in[1];   // [T,N,2]
  const float* W_W     = (const float*)d_in[2];   // [128,512]
  const float* b_W     = (const float*)d_in[3];   // [512]
  const float* U_f1    = (const float*)d_in[4];   // [256,128]
  const float* U_f2    = (const float*)d_in[5];   // [256,128]
  const float* U_iuo   = (const float*)d_in[6];   // [256,384]

  float* resH = (float*)d_out;                          // [T,N,128]
  float* resC = resH + (size_t)T_STEPS * NN * DD;       // [T,N,128]

  unsigned short* packed = (unsigned short*)d_ws;       // 229376 bf16 = 448 KB

  // one-time weight pack into jt-grouped WMMA B-fragment order
  pack_weights_kernel<<<(TOTAL_PACKED + 255) / 256, 256, 0, stream>>>(
      W_W, U_f1, U_f2, U_iuo, packed);

  // 64 dependent steps, ordered by the stream
  for (int t = 0; t < T_STEPS; ++t) {
    const float* prevH = (t > 0) ? (resH + (size_t)(t - 1) * NN * DD) : nullptr;
    const float* prevC = (t > 0) ? (resC + (size_t)(t - 1) * NN * DD) : nullptr;
    tree_lstm_step_kernel<<<NN / (16 * 16), 512, 0, stream>>>(
        tensor + (size_t)t * NN * DD,
        indices + (size_t)t * NN * 2,
        packed, b_W,
        prevH, prevC,
        resH + (size_t)t * NN * DD,
        resC + (size_t)t * NN * DD);
  }
}
